// MultiHeadAttention_68685116998345
// MI455X (gfx1250) — compile-verified, matrix operand order checked by reading, not run on hardware
//
#include <hip/hip_runtime.h>
#include <hip/hip_bf16.h>
#include <math.h>

#define D_MODEL 1024
#define NUM_HEADS 16
#define D_K 64
#define BATCH 4
#define SEQ 2048
#define BS (BATCH * SEQ) /* 8192 */

typedef __bf16 v16bf __attribute__((ext_vector_type(16)));
typedef float  v8f   __attribute__((ext_vector_type(8)));
typedef unsigned int v4u __attribute__((ext_vector_type(4)));
typedef int v8i __attribute__((ext_vector_type(8)));
typedef int v4i __attribute__((ext_vector_type(4)));

union FragAB { v16bf v; unsigned int u[8]; };
union FragC  { v8f   v; float        f[8]; };
union Pack4  { __bf16 h[4]; unsigned long long q; };
union Pack2  { __bf16 h[2]; unsigned int u; };

#ifdef __has_builtin
#if __has_builtin(__builtin_amdgcn_tensor_load_to_lds) && \
    __has_builtin(__builtin_amdgcn_s_wait_tensorcnt)
#define HAVE_TDM 1
#endif
#endif
#ifndef HAVE_TDM
#define HAVE_TDM 0
#endif

// CDNA5 16-bit A-fragment K index for pair p (0..7) and lane-half (0..1).
// Lanes 0-15: VGPR0..3 -> K=0..7, VGPR4..7 -> K=16..23; lanes 16-31: +8.
__device__ __forceinline__ int frag_k(int p, int half) {
  int base = (p < 4) ? (p * 2) : (16 + (p - 4) * 2);
  return base + half * 8;
}

__device__ __forceinline__ unsigned int lds_pair(const __bf16* p) {
  return *(const unsigned int*)p;  // frag_k is always even -> 4B aligned
}

// Async global->LDS 16-byte copy (GLOBAL_LOAD_ASYNC_TO_LDS_B128, ASYNCcnt).
__device__ __forceinline__ void async_copy_b128(unsigned lds_addr,
                                                const void* gaddr) {
  asm volatile("global_load_async_to_lds_b128 %0, %1, off"
               :: "v"(lds_addr), "v"(gaddr)
               : "memory");
}
__device__ __forceinline__ void wait_asynccnt0() {
  asm volatile("s_wait_asynccnt 0x0" ::: "memory");
}
// Low 32 bits of a flat pointer to a __shared__ object == LDS byte address.
__device__ __forceinline__ unsigned lds_addr_of(const void* p) {
  return (unsigned)(unsigned long long)p;
}

// ---------------------------------------------------------------------------
// Kernel 1: fused per-head Q/K/V projection.
// grid (BS/64, H), block 384 = 12 waves. wave%3 selects {q,k,v}, wave/3 the
// 16-row slice. LDS stages bf16 X tiles and W tiles (transposed to [n][k]).
// ---------------------------------------------------------------------------
__global__ __launch_bounds__(384)
void qkv_proj_kernel(const float* __restrict__ q_in, const float* __restrict__ k_in,
                     const float* __restrict__ v_in,
                     const float* __restrict__ Wq, const float* __restrict__ bq,
                     const float* __restrict__ Wk, const float* __restrict__ bk,
                     const float* __restrict__ Wv, const float* __restrict__ bv,
                     __bf16* __restrict__ Qb, __bf16* __restrict__ Kb,
                     __bf16* __restrict__ Vb) {
  __shared__ alignas(16) __bf16 lx[3][64][32];   // [mat][row][k]   12 KB
  __shared__ alignas(16) __bf16 lw[3][64][32];   // [mat][n=dk][k]  12 KB

  const int tid = threadIdx.x;
  const int wave = tid >> 5, lane = tid & 31;
  const int msel = wave % 3;      // 0=q,1=k,2=v
  const int rowtile = wave / 3;   // 0..3
  const int h = blockIdx.y;
  const int row0 = blockIdx.x * 64;
  const int m = lane & 15, half = lane >> 4;

  FragC acc[4];
  #pragma unroll
  for (int nt = 0; nt < 4; ++nt)
    #pragma unroll
    for (int r = 0; r < 8; ++r) acc[nt].f[r] = 0.0f;

  for (int kk0 = 0; kk0 < D_MODEL; kk0 += 32) {
    // X tiles: 3*64*32 f32 -> bf16, vectorized (float4 load, b64 LDS store)
    for (int i = tid; i < 1536; i += 384) {
      int mm = i / 512, rem = i % 512;
      int r = rem >> 3, c = (rem & 7) * 4;
      const float* X = (mm == 0) ? q_in : (mm == 1) ? k_in : v_in;
      float4 xv = *(const float4*)&X[(size_t)(row0 + r) * D_MODEL + kk0 + c];
      Pack4 pk;
      pk.h[0] = (__bf16)xv.x; pk.h[1] = (__bf16)xv.y;
      pk.h[2] = (__bf16)xv.z; pk.h[3] = (__bf16)xv.w;
      *(unsigned long long*)&lx[mm][r][c] = pk.q;
    }
    // W tiles transposed to [n][k]: vectorize along n (contiguous in memory)
    for (int i = tid; i < 1536; i += 384) {
      int mm = i / 512, rem = i % 512;
      int c = rem >> 4, n = (rem & 15) * 4;
      const float* W = (mm == 0) ? Wq : (mm == 1) ? Wk : Wv;
      float4 wv =
          *(const float4*)&W[(size_t)h * D_MODEL * D_K + (size_t)(kk0 + c) * D_K + n];
      lw[mm][n + 0][c] = (__bf16)wv.x;
      lw[mm][n + 1][c] = (__bf16)wv.y;
      lw[mm][n + 2][c] = (__bf16)wv.z;
      lw[mm][n + 3][c] = (__bf16)wv.w;
    }
    __syncthreads();

    FragAB a;
    #pragma unroll
    for (int p = 0; p < 8; ++p)
      a.u[p] = lds_pair(&lx[msel][rowtile * 16 + m][frag_k(p, half)]);

    #pragma unroll
    for (int nt = 0; nt < 4; ++nt) {
      FragAB b;
      #pragma unroll
      for (int p = 0; p < 8; ++p)
        b.u[p] = lds_pair(&lw[msel][nt * 16 + m][frag_k(p, half)]);
      acc[nt].v = __builtin_amdgcn_wmma_f32_16x16x32_bf16(
          false, a.v, false, b.v, (short)0, acc[nt].v, false, false);
    }
    __syncthreads();
  }

  const float* bias = (msel == 0) ? bq : (msel == 1) ? bk : bv;
  __bf16* outp = (msel == 0) ? Qb : (msel == 1) ? Kb : Vb;

  #pragma unroll
  for (int nt = 0; nt < 4; ++nt) {
    int n = nt * 16 + m;
    float bval = bias[h * D_K + n];
    #pragma unroll
    for (int r = 0; r < 8; ++r) {
      int row_g = row0 + rowtile * 16 + r + 8 * half;  // index into B*S
      int b = row_g >> 11;          // / SEQ
      int s = row_g & (SEQ - 1);
      outp[(((size_t)b * NUM_HEADS + h) * SEQ + s) * D_K + n] =
          (__bf16)(acc[nt].f[r] + bval);
    }
  }
}

// ---------------------------------------------------------------------------
// Kernel 2: flash attention per (b,h).
// grid (SEQ/64, B*H), block 128 = 4 waves; each wave owns 16 query rows and
// streams 64-key blocks with online softmax. K blocks are staged by the
// Tensor Data Mover (one tensor_load_to_lds per workgroup, TENSORcnt) when
// available, else async global->LDS b128 copies. V is transposed in-flight.
// P bounces through wave-private LDS to convert C-layout into A-layout.
// ---------------------------------------------------------------------------
__global__ __launch_bounds__(128)
void attention_kernel(const __bf16* __restrict__ Qb, const __bf16* __restrict__ Kb,
                      const __bf16* __restrict__ Vb, __bf16* __restrict__ Cc) {
  __shared__ alignas(16) __bf16 lk[64][64];     // [key][dk]      8 KB
  __shared__ alignas(16) __bf16 lv[64][64];     // [dk][key]      8 KB
  __shared__ alignas(16) __bf16 lp[4][16][64];  // [wave][qrow][key] 8 KB

  const int tid = threadIdx.x;
  const int wave = tid >> 5, lane = tid & 31;
  const int m = lane & 15, half = lane >> 4;
  const int bh = blockIdx.y;
  const int h = bh % NUM_HEADS, b = bh / NUM_HEADS;
  const int q0 = blockIdx.x * 64 + wave * 16;

  // Q fragments (A layout), fixed for the whole key loop.
  FragAB aQ[2];
  {
    const unsigned int* qrow =
        (const unsigned int*)(Qb + ((size_t)bh * SEQ + q0 + m) * D_K);
    #pragma unroll
    for (int s = 0; s < 2; ++s)
      #pragma unroll
      for (int p = 0; p < 8; ++p)
        aQ[s].u[p] = qrow[(frag_k(p, half) + s * 32) >> 1];
  }

  FragC o[4];
  float m_r[8], l_r[8];
  #pragma unroll
  for (int r = 0; r < 8; ++r) { m_r[r] = -INFINITY; l_r[r] = 0.0f; }
  #pragma unroll
  for (int nt = 0; nt < 4; ++nt)
    #pragma unroll
    for (int r = 0; r < 8; ++r) o[nt].f[r] = 0.0f;

  const unsigned lk_base = lds_addr_of(&lk[0][0]);

  for (int kb0 = 0; kb0 < SEQ; kb0 += 64) {
#if HAVE_TDM
    // --- stage K block via Tensor Data Mover: one 64(key) x 64(dk) bf16
    // tile. D# group0: count=1 | lds_addr | global_addr | type=2.
    // group1: data_size=2B, tensor=[2048,64], tile=[64,64], dim0_stride=64.
    if (wave == 0) {
      const unsigned long long ga =
          (unsigned long long)(const void*)(Kb + ((size_t)bh * SEQ + kb0) * D_K);
      v4u g0;
      g0.x = 1u;                                  // count=1 (valid descriptor)
      g0.y = lk_base;                             // lds_addr
      g0.z = (unsigned)ga;                        // global_addr[31:0]
      g0.w = (unsigned)((ga >> 32) & 0x01FFFFFFu) | (2u << 30);  // [56:32]|type=2
      v8i g1;
      g1[0] = (int)(1u << 16);       // workgroup_mask=0, data_size=1 (2 bytes)
      g1[1] = (int)(64u << 16);      // tensor_dim0 = 64 (bits 63:48)
      g1[2] = (int)(2048u << 16);    // tensor_dim1 = 2048 (bits 95:80)
      g1[3] = (int)(64u << 16);      // tile_dim0 = 64 (bits 127:112)
      g1[4] = 64;                    // tile_dim1 = 64, tile_dim2 = 0
      g1[5] = 64;                    // tensor_dim0_stride = 64
      g1[6] = 0;                     // stride hi, tensor_dim1_stride lo
      g1[7] = 0;                     // tensor_dim1_stride hi
      v4i g2 = {0, 0, 0, 0}, g3 = {0, 0, 0, 0};   // 2D: groups 2/3 unused
      v8i g4 = {0, 0, 0, 0, 0, 0, 0, 0};          // extra group (clang-23 6-arg form)
      __builtin_amdgcn_tensor_load_to_lds(g0, g1, g2, g3, g4, 0);
      __builtin_amdgcn_s_wait_tensorcnt(0);
    }
#else
    // --- stage K block via async global->LDS (8 KB = 128 lanes x 4 x b128)
    {
      const char* gk = (const char*)(Kb + ((size_t)bh * SEQ + kb0) * D_K);
      #pragma unroll
      for (int it = 0; it < 4; ++it)
        async_copy_b128(lk_base + (unsigned)(tid * 16 + it * 2048),
                        gk + tid * 16 + it * 2048);
    }
#endif
    // --- stage V block transposed: [key][dk] -> lv[dk][key]
    {
      const unsigned int* vsrc =
          (const unsigned int*)(Vb + ((size_t)bh * SEQ + kb0) * D_K);
      for (int i = tid; i < 64 * 32; i += 128) {
        int r = i >> 5, c2 = (i & 31) * 2;
        Pack2 pv; pv.u = vsrc[i];
        lv[c2 + 0][r] = pv.h[0];
        lv[c2 + 1][r] = pv.h[1];
      }
    }
    // prefetch next V block while we compute on this one
    if (kb0 + 64 < SEQ)
      __builtin_prefetch(Vb + ((size_t)bh * SEQ + kb0 + 64) * D_K + tid * 32, 0, 0);

#if !HAVE_TDM
    wait_asynccnt0();
#endif
    __syncthreads();

    // S = Q * K^T  (contraction over dk=64: two 32-wide WMMA steps)
    FragC sc[4];
    #pragma unroll
    for (int nt = 0; nt < 4; ++nt)
      #pragma unroll
      for (int r = 0; r < 8; ++r) sc[nt].f[r] = 0.0f;

    #pragma unroll
    for (int s = 0; s < 2; ++s) {
      #pragma unroll
      for (int nt = 0; nt < 4; ++nt) {
        FragAB bfr;
        #pragma unroll
        for (int p = 0; p < 8; ++p)
          bfr.u[p] = lds_pair(&lk[nt * 16 + m][frag_k(p, half) + s * 32]);
        sc[nt].v = __builtin_amdgcn_wmma_f32_16x16x32_bf16(
            false, aQ[s].v, false, bfr.v, (short)0, sc[nt].v, false, false);
      }
    }

    // online softmax; rows live in 16-lane groups, so xor-shuffles of 1,2,4,8
    // stay within the row group.
    float alpha[8];
    #pragma unroll
    for (int r = 0; r < 8; ++r) {
      float mx = -INFINITY;
      #pragma unroll
      for (int nt = 0; nt < 4; ++nt) {
        sc[nt].f[r] *= 0.125f;  // 1/sqrt(64)
        mx = fmaxf(mx, sc[nt].f[r]);
      }
      #pragma unroll
      for (int off = 8; off >= 1; off >>= 1)
        mx = fmaxf(mx, __shfl_xor(mx, off, 32));
      mx = fmaxf(mx, m_r[r]);
      alpha[r] = __expf(m_r[r] - mx);
      m_r[r] = mx;
    }
    #pragma unroll
    for (int r = 0; r < 8; ++r) {
      float sum = 0.0f;
      #pragma unroll
      for (int nt = 0; nt < 4; ++nt) {
        float e = __expf(sc[nt].f[r] - m_r[r]);
        sc[nt].f[r] = e;
        sum += e;
      }
      #pragma unroll
      for (int off = 8; off >= 1; off >>= 1)
        sum += __shfl_xor(sum, off, 32);
      l_r[r] = l_r[r] * alpha[r] + sum;
      #pragma unroll
      for (int nt = 0; nt < 4; ++nt) o[nt].f[r] *= alpha[r];
    }

    // spill P (C-layout) to wave-private LDS, reload in A layout
    #pragma unroll
    for (int nt = 0; nt < 4; ++nt)
      #pragma unroll
      for (int r = 0; r < 8; ++r)
        lp[wave][r + 8 * half][nt * 16 + m] = (__bf16)sc[nt].f[r];
    asm volatile("s_wait_dscnt 0x0" ::: "memory");

    // O += P * V  (contraction over 64 keys: two 32-wide WMMA steps)
    #pragma unroll
    for (int s = 0; s < 2; ++s) {
      FragAB aP;
      #pragma unroll
      for (int p = 0; p < 8; ++p)
        aP.u[p] = lds_pair(&lp[wave][m][frag_k(p, half) + s * 32]);
      #pragma unroll
      for (int nt = 0; nt < 4; ++nt) {
        FragAB bV;
        #pragma unroll
        for (int p = 0; p < 8; ++p)
          bV.u[p] = lds_pair(&lv[nt * 16 + m][frag_k(p, half) + s * 32]);
        o[nt].v = __builtin_amdgcn_wmma_f32_16x16x32_bf16(
            false, aP.v, false, bV.v, (short)0, o[nt].v, false, false);
      }
    }
    __syncthreads();
  }

  // normalize, write concat-heads [B, S, H*64] as bf16
  #pragma unroll
  for (int r = 0; r < 8; ++r) {
    float inv = 1.0f / l_r[r];
    int s_g = q0 + r + 8 * half;
    #pragma unroll
    for (int nt = 0; nt < 4; ++nt)
      Cc[((size_t)b * SEQ + s_g) * D_MODEL + h * D_K + nt * 16 + m] =
          (__bf16)(o[nt].f[r] * inv);
  }
}

// ---------------------------------------------------------------------------
// Kernel 3: output projection  out = concat @ Wo + bo.
// grid (BS/32, D_MODEL/256), block 256 = 8 waves (2 row-tiles x 4 col-tiles).
// The bf16 A tile is staged with async global->LDS copies; the f32 Wo tile
// needs conversion, so it uses vectorized float4 loads.
// ---------------------------------------------------------------------------
__global__ __launch_bounds__(256)
void out_proj_kernel(const __bf16* __restrict__ Cc, const float* __restrict__ Wo,
                     const float* __restrict__ bo, float* __restrict__ out) {
  __shared__ alignas(16) __bf16 la[32][32];    // A tile, 2 KB
  __shared__ alignas(16) __bf16 lw[256][32];   // Wo tile transposed [n][k], 16 KB

  const int tid = threadIdx.x;
  const int wave = tid >> 5, lane = tid & 31;
  const int m = lane & 15, half = lane >> 4;
  const int rowtile = wave >> 2, coltile = wave & 3;
  const int row0 = blockIdx.x * 32;
  const int nbase = blockIdx.y * 256;

  const unsigned la_base = lds_addr_of(&la[0][0]);

  FragC acc[4];
  #pragma unroll
  for (int nt = 0; nt < 4; ++nt)
    #pragma unroll
    for (int r = 0; r < 8; ++r) acc[nt].f[r] = 0.0f;

  for (int kk0 = 0; kk0 < D_MODEL; kk0 += 32) {
    // A tile: 32 rows x 64B, async b128 copies (128 lanes x 16B)
    if (tid < 128) {
      int r = tid >> 2, c4 = tid & 3;
      async_copy_b128(la_base + (unsigned)(r * 64 + c4 * 16),
                      (const char*)&Cc[(size_t)(row0 + r) * D_MODEL + kk0] + c4 * 16);
    }
    // Wo tile transposed to [n][k], vectorized along n
    for (int i = tid; i < 2048; i += 256) {
      int c = i >> 6, n = (i & 63) * 4;
      float4 wv = *(const float4*)&Wo[(size_t)(kk0 + c) * D_MODEL + nbase + n];
      lw[n + 0][c] = (__bf16)wv.x;
      lw[n + 1][c] = (__bf16)wv.y;
      lw[n + 2][c] = (__bf16)wv.z;
      lw[n + 3][c] = (__bf16)wv.w;
    }
    wait_asynccnt0();
    __syncthreads();

    FragAB a;
    #pragma unroll
    for (int p = 0; p < 8; ++p)
      a.u[p] = lds_pair(&la[rowtile * 16 + m][frag_k(p, half)]);

    #pragma unroll
    for (int nt = 0; nt < 4; ++nt) {
      FragAB b;
      #pragma unroll
      for (int p = 0; p < 8; ++p)
        b.u[p] = lds_pair(&lw[coltile * 64 + nt * 16 + m][frag_k(p, half)]);
      acc[nt].v = __builtin_amdgcn_wmma_f32_16x16x32_bf16(
          false, a.v, false, b.v, (short)0, acc[nt].v, false, false);
    }
    __syncthreads();
  }

  #pragma unroll
  for (int nt = 0; nt < 4; ++nt) {
    int col = nbase + coltile * 64 + nt * 16 + m;
    float bval = bo[col];
    #pragma unroll
    for (int r = 0; r < 8; ++r) {
      int row = row0 + rowtile * 16 + r + 8 * half;
      out[(size_t)row * D_MODEL + col] = acc[nt].f[r] + bval;
    }
  }
}

// ---------------------------------------------------------------------------
extern "C" void kernel_launch(void* const* d_in, const int* in_sizes, int n_in,
                              void* d_out, int out_size, void* d_ws, size_t ws_size,
                              hipStream_t stream) {
  const float* query = (const float*)d_in[0];
  const float* key_  = (const float*)d_in[1];
  const float* value = (const float*)d_in[2];
  const float* Wq = (const float*)d_in[3];
  const float* bq = (const float*)d_in[4];
  const float* Wk = (const float*)d_in[5];
  const float* bk = (const float*)d_in[6];
  const float* Wv = (const float*)d_in[7];
  const float* bv = (const float*)d_in[8];
  const float* Wo = (const float*)d_in[9];
  const float* bo = (const float*)d_in[10];
  float* out = (float*)d_out;

  // Workspace: Qb | Kb | Vb | Cc, each [8192 x 1024] bf16 = 16 MiB (64 MiB total)
  const size_t elems = (size_t)BS * D_MODEL;
  __bf16* Qb = (__bf16*)d_ws;
  __bf16* Kb = Qb + elems;
  __bf16* Vb = Kb + elems;
  __bf16* Cc = Vb + elems;

  qkv_proj_kernel<<<dim3(BS / 64, NUM_HEADS), 384, 0, stream>>>(
      query, key_, value, Wq, bq, Wk, bk, Wv, bv, Qb, Kb, Vb);
  attention_kernel<<<dim3(SEQ / 64, BATCH * NUM_HEADS), 128, 0, stream>>>(
      Qb, Kb, Vb, Cc);
  out_proj_kernel<<<dim3(BS / 32, D_MODEL / 256), 256, 0, stream>>>(
      Cc, Wo, bo, out);
}
